// GVP_embedding_70239895159026
// MI455X (gfx1250) — compile-verified, matrix operand order checked by reading, not run on hardware
//
#include <hip/hip_runtime.h>
#include <hip/hip_bf16.h>

// ---------------------------------------------------------------------------
// GVP-GNN for MI455X (gfx1250, wave32). All large scalar matmuls run through
// v_wmma_f32_16x16x32_f16 (f16 inputs, f32 accumulate). Small 3-component
// vector channel mixes + norms/LN/sigmoid run in f32 VALU with LDS staging.
// ---------------------------------------------------------------------------

typedef _Float16 h16;
typedef __attribute__((ext_vector_type(8)))  _Float16 v8h;
typedef __attribute__((ext_vector_type(16))) _Float16 v16h;
typedef __attribute__((ext_vector_type(8)))  float    v8f;

#define NS 128
#define NV 16
#define SE 32

__device__ inline float wsum(float v) {
  #pragma unroll
  for (int o = 16; o > 0; o >>= 1) v += __shfl_xor(v, o, 32);
  return v;
}
__device__ inline float sigm(float x) { return 1.f / (1.f + __expf(-x)); }

// ---------------------------------------------------------------------------
// Weight repack: f32 row-major W[K][N] -> f16 packed B-lane layout.
// Tile (kt,nt): 32 lanes x 16 halves, element (2i+h) of lane L is
// B[kt*32 + kbase(i,h,L)][nt*16 + (L&15)] per the CDNA5 16-bit B layout.
// grid = (Kp/32, N/16), block = 32.
// ---------------------------------------------------------------------------
__global__ void __launch_bounds__(32)
pack_w(const float* __restrict__ W, h16* __restrict__ Bp, int K, int N) {
  const int kt = blockIdx.x, nt = blockIdx.y, lane = threadIdx.x;
  const int Ntiles = gridDim.y;
  const int n = nt * 16 + (lane & 15);
  const int hi8 = (lane >= 16) ? 8 : 0;
  union { v16h v; h16 a[16]; } u;
  #pragma unroll
  for (int i = 0; i < 8; ++i) {
    const int kb = (i < 4 ? 2 * i : 16 + 2 * (i - 4)) + hi8;
    #pragma unroll
    for (int hh = 0; hh < 2; ++hh) {
      const int k = kt * 32 + kb + hh;
      u.a[2 * i + hh] = (h16)((k < K) ? W[(size_t)k * N + n] : 0.f);
    }
  }
  *(v16h*)(Bp + (((size_t)kt * Ntiles + nt) * 32 + lane) * 16) = u.v;
}

// ---------------------------------------------------------------------------
// Generic WMMA GEMM: C[M x N] = act(A_f16[M x Kp] * Wpacked + bias)
// One wave per 16 x (16*NT) output tile strip.
// ---------------------------------------------------------------------------
template <int NT>
__global__ void __launch_bounds__(32)
gemm_wmma(const h16* __restrict__ A, const h16* __restrict__ Bp,
          const float* __restrict__ bias, float* __restrict__ C,
          int M, int Kp, int N, int relu_act) {
  const int lane = threadIdx.x;
  const int mt = blockIdx.x, ng = blockIdx.y;
  const int Ntiles = N >> 4;
  const int ktiles = Kp >> 5;
  const int m = lane & 15;
  const size_t arow = (size_t)(mt * 16 + m) * Kp + ((lane >> 4) << 3);
  v8f acc[NT];
  #pragma unroll
  for (int t = 0; t < NT; ++t) { v8f z = {0,0,0,0,0,0,0,0}; acc[t] = z; }
  for (int kt = 0; kt < ktiles; ++kt) {
    const h16* ap = A + arow + kt * 32;
    v8h alo = *(const v8h*)(ap);
    v8h ahi = *(const v8h*)(ap + 16);
    v16h a = __builtin_shufflevector(alo, ahi, 0,1,2,3,4,5,6,7,8,9,10,11,12,13,14,15);
    #pragma unroll
    for (int t = 0; t < NT; ++t) {
      const int nt = ng * NT + t;
      v16h b = *(const v16h*)(Bp + (((size_t)kt * Ntiles + nt) * 32 + lane) * 16);
      acc[t] = __builtin_amdgcn_wmma_f32_16x16x32_f16(
          false, a, false, b, (short)0, acc[t], false, false);
    }
  }
  const int mo = (lane < 16) ? 0 : 8;
  const int nc = lane & 15;
  #pragma unroll
  for (int t = 0; t < NT; ++t) {
    const int col = (ng * NT + t) * 16 + nc;
    const float bv = bias ? bias[col] : 0.f;
    #pragma unroll
    for (int j = 0; j < 8; ++j) {
      const int row = mt * 16 + mo + j;
      float val = acc[t][j] + bv;
      if (relu_act) val = fmaxf(val, 0.f);
      if (row < M) C[(size_t)row * N + col] = val;
    }
  }
}

__global__ void zerof(float* p, size_t n) {
  size_t i = (size_t)blockIdx.x * blockDim.x + threadIdx.x;
  if (i < n) p[i] = 0.f;
}

// ---------------------------------------------------------------------------
// Node embedding: hs16 = [node_s(6), W_s[seq](10)] -> LN(16); v RMS-norm;
// vh = v*wh (3x16); A = f16[s(16), |vh|(16)]; store vh. Wave per node.
// ---------------------------------------------------------------------------
__global__ void __launch_bounds__(256)
embed_nodes(const float* __restrict__ node_s, const float* __restrict__ node_v,
            const int* __restrict__ seq, const float* __restrict__ Ws,
            const float* __restrict__ lng, const float* __restrict__ lnb,
            const float* __restrict__ wh, h16* __restrict__ A,
            float* __restrict__ vhn, int N) {
  const int lane = threadIdx.x & 31, wid = threadIdx.x >> 5;
  const int n = blockIdx.x * 8 + wid;
  if (n >= N) return;
  float s = 0.f;
  if (lane < 16)
    s = (lane < 6) ? node_s[(size_t)n * 6 + lane] : Ws[seq[n] * 10 + (lane - 6)];
  const float mu = wsum(lane < 16 ? s : 0.f) * (1.f / 16.f);
  const float d = (lane < 16) ? (s - mu) : 0.f;
  const float var = wsum(d * d) * (1.f / 16.f);
  if (lane < 16) s = d * rsqrtf(var + 1e-5f) * lng[lane] + lnb[lane];
  float v[9];
  #pragma unroll
  for (int i = 0; i < 9; ++i) v[i] = node_v[(size_t)n * 9 + i];
  float m = 0.f;
  #pragma unroll
  for (int i = 0; i < 3; ++i) {
    float q = v[i*3]*v[i*3] + v[i*3+1]*v[i*3+1] + v[i*3+2]*v[i*3+2];
    m += fmaxf(q, 1e-8f);
  }
  const float sc = rsqrtf(m * (1.f / 3.f));
  float out = s;
  if (lane >= 16) {
    const int h = lane - 16;
    float a0 = 0, a1 = 0, a2 = 0;
    #pragma unroll
    for (int i = 0; i < 3; ++i) {
      const float w = wh[i * 16 + h];
      a0 += v[i*3+0] * w; a1 += v[i*3+1] * w; a2 += v[i*3+2] * w;
    }
    a0 *= sc; a1 *= sc; a2 *= sc;
    vhn[((size_t)n * 16 + h) * 3 + 0] = a0;
    vhn[((size_t)n * 16 + h) * 3 + 1] = a1;
    vhn[((size_t)n * 16 + h) * 3 + 2] = a2;
    out = sqrtf(fmaxf(a0*a0 + a1*a1 + a2*a2, 1e-8f));
  }
  A[(size_t)n * 32 + lane] = (h16)out;
}

// vo[n,o,c] = sum_h vh[n,h,c]*wv[h,o]  (thread per (n,o))
__global__ void node_wv(const float* __restrict__ vh, const float* __restrict__ wv,
                        float* __restrict__ hv, int N) {
  const int t = blockIdx.x * blockDim.x + threadIdx.x;
  if (t >= N * 16) return;
  const int n = t >> 4, o = t & 15;
  float a0 = 0, a1 = 0, a2 = 0;
  for (int h = 0; h < 16; ++h) {
    const float w = wv[h * 16 + o];
    a0 += vh[((size_t)n*16+h)*3+0] * w;
    a1 += vh[((size_t)n*16+h)*3+1] * w;
    a2 += vh[((size_t)n*16+h)*3+2] * w;
  }
  hv[((size_t)n*16+o)*3+0] = a0;
  hv[((size_t)n*16+o)*3+1] = a1;
  hv[((size_t)n*16+o)*3+2] = a2;
}

// ---------------------------------------------------------------------------
// Edge embedding: LN(32); ev RMS; vh=ev*wh00; A=f16[s(32),|vh|,pad to 64];
// ev' = vh*wv00. Wave per edge.
// ---------------------------------------------------------------------------
__global__ void __launch_bounds__(256)
embed_edges(const float* __restrict__ es_in, const float* __restrict__ ev_in,
            const float* __restrict__ lng, const float* __restrict__ lnb,
            const float* __restrict__ wh, const float* __restrict__ wv,
            h16* __restrict__ A, float* __restrict__ evout, int E) {
  const int lane = threadIdx.x & 31, wid = threadIdx.x >> 5;
  const int e = blockIdx.x * 8 + wid;
  if (e >= E) return;
  float s = es_in[(size_t)e * 32 + lane];
  const float mu = wsum(s) * (1.f / 32.f);
  const float d = s - mu;
  const float var = wsum(d * d) * (1.f / 32.f);
  s = d * rsqrtf(var + 1e-5f) * lng[lane] + lnb[lane];
  const float v0 = ev_in[(size_t)e*3+0], v1 = ev_in[(size_t)e*3+1], v2 = ev_in[(size_t)e*3+2];
  const float sc = rsqrtf(fmaxf(v0*v0 + v1*v1 + v2*v2, 1e-8f));
  const float w00 = wh[0];
  const float h0 = v0*sc*w00, h1 = v1*sc*w00, h2 = v2*sc*w00;
  const float nrm = sqrtf(fmaxf(h0*h0 + h1*h1 + h2*h2, 1e-8f));
  if (lane == 0) {
    const float wv00 = wv[0];
    evout[(size_t)e*3+0] = h0*wv00; evout[(size_t)e*3+1] = h1*wv00; evout[(size_t)e*3+2] = h2*wv00;
  }
  A[(size_t)e*64 + lane]      = (h16)s;
  A[(size_t)e*64 + 32 + lane] = (h16)(lane == 0 ? nrm : 0.f);
}

// ---------------------------------------------------------------------------
// conv0 pre: gather [hv_src, ev, hv_dst] (33x3) -> vh=mv*wh0 (33x33), norms;
// stage A0 = f16[hs_src(128), es(32), hs_dst(128), |vh|(33), pad->352].
// ---------------------------------------------------------------------------
__global__ void __launch_bounds__(256)
conv_pre(const float* __restrict__ hs, const float* __restrict__ es,
         const float* __restrict__ hv, const float* __restrict__ ev,
         const int* __restrict__ eidx, const float* __restrict__ wh0,
         float* __restrict__ vh33, h16* __restrict__ A0, int E) {
  __shared__ float smv[8][33][3];
  __shared__ float ssc[8][324];
  const int lane = threadIdx.x & 31, wid = threadIdx.x >> 5;
  const int e = blockIdx.x * 8 + wid;
  const bool ok = (e < E);
  int src = 0, dst = 0;
  if (ok) { src = eidx[e]; dst = eidx[E + e]; }
  if (ok) for (int idx = lane; idx < 99; idx += 32) {
    const int i = idx / 3, c = idx - 3 * i;
    float val;
    if (i < 16)       val = hv[((size_t)src*16 + i) * 3 + c];
    else if (i == 16) val = ev[(size_t)e*3 + c];
    else              val = hv[((size_t)dst*16 + (i-17)) * 3 + c];
    smv[wid][i][c] = val;
  }
  if (ok) for (int idx = lane; idx < 288; idx += 32) {
    float val;
    if (idx < 128)      val = hs[(size_t)src*128 + idx];
    else if (idx < 160) val = es[(size_t)e*32 + (idx-128)];
    else                val = hs[(size_t)dst*128 + (idx-160)];
    ssc[wid][idx] = val;
  }
  __syncthreads();
  if (ok) for (int h = lane; h < 33; h += 32) {
    float a0 = 0, a1 = 0, a2 = 0;
    for (int i = 0; i < 33; ++i) {
      const float w = wh0[i * 33 + h];
      a0 += smv[wid][i][0]*w; a1 += smv[wid][i][1]*w; a2 += smv[wid][i][2]*w;
    }
    vh33[((size_t)e*33 + h)*3 + 0] = a0;
    vh33[((size_t)e*33 + h)*3 + 1] = a1;
    vh33[((size_t)e*33 + h)*3 + 2] = a2;
    ssc[wid][288 + h] = sqrtf(fmaxf(a0*a0 + a1*a1 + a2*a2, 1e-8f));
  }
  __syncthreads();
  if (ok) for (int idx = lane; idx < 352; idx += 32)
    A0[(size_t)e*352 + idx] = (h16)((idx < 321) ? ssc[wid][idx] : 0.f);
}

// ---------------------------------------------------------------------------
// conv mid: vo = sigmoid-gate(vh_in*wv); vh_next = vo*wh_next (16x16);
// stage A1 = f16[ms(128), |vh_next|(16), pad->160]. In-place safe if strides equal.
// ---------------------------------------------------------------------------
__global__ void __launch_bounds__(256)
conv_mid_post(const float* __restrict__ vh_in, int Hin,
              const float* __restrict__ wv, const float* __restrict__ wh_next,
              const float* __restrict__ ms, float* __restrict__ mv_out,
              float* __restrict__ vh_next, h16* __restrict__ A1, int E) {
  __shared__ float sv[8][100];
  __shared__ float sm[8][48];
  const int lane = threadIdx.x & 31, wid = threadIdx.x >> 5;
  const int e = blockIdx.x * 8 + wid;
  const bool ok = (e < E);
  if (ok) for (int idx = lane; idx < Hin * 3; idx += 32)
    sv[wid][idx] = vh_in[(size_t)e * Hin * 3 + idx];
  __syncthreads();
  if (ok && lane < 16) {
    const int o = lane;
    float m0 = 0, m1 = 0, m2 = 0;
    for (int h = 0; h < Hin; ++h) {
      const float w = wv[h * 16 + o];
      m0 += sv[wid][h*3+0]*w; m1 += sv[wid][h*3+1]*w; m2 += sv[wid][h*3+2]*w;
    }
    const float g = sigm(sqrtf(fmaxf(m0*m0 + m1*m1 + m2*m2, 1e-8f)));
    m0 *= g; m1 *= g; m2 *= g;
    sm[wid][o*3+0] = m0; sm[wid][o*3+1] = m1; sm[wid][o*3+2] = m2;
    mv_out[((size_t)e*16 + o)*3 + 0] = m0;
    mv_out[((size_t)e*16 + o)*3 + 1] = m1;
    mv_out[((size_t)e*16 + o)*3 + 2] = m2;
  }
  __syncthreads();
  if (ok && lane < 16) {
    const int h = lane;
    float a0 = 0, a1 = 0, a2 = 0;
    for (int o = 0; o < 16; ++o) {
      const float w = wh_next[o * 16 + h];
      a0 += sm[wid][o*3+0]*w; a1 += sm[wid][o*3+1]*w; a2 += sm[wid][o*3+2]*w;
    }
    vh_next[((size_t)e*16 + h)*3 + 0] = a0;
    vh_next[((size_t)e*16 + h)*3 + 1] = a1;
    vh_next[((size_t)e*16 + h)*3 + 2] = a2;
    A1[(size_t)e*160 + 128 + h]    = (h16)sqrtf(fmaxf(a0*a0 + a1*a1 + a2*a2, 1e-8f));
    A1[(size_t)e*160 + 144 + lane] = (h16)0.f;
  }
  if (ok) for (int idx = lane; idx < 128; idx += 32)
    A1[(size_t)e*160 + idx] = (h16)ms[(size_t)e*128 + idx];
}

// conv final: mv3 = vh_in*wv2 (no gate); scatter-add ms/mv3/count by dst.
__global__ void __launch_bounds__(256)
conv_final(const float* __restrict__ vh_in, const float* __restrict__ wv2,
           const float* __restrict__ ms, const int* __restrict__ eidx,
           float* __restrict__ sum_s, float* __restrict__ sum_v,
           float* __restrict__ cnt, int E) {
  __shared__ float sv[8][48];
  const int lane = threadIdx.x & 31, wid = threadIdx.x >> 5;
  const int e = blockIdx.x * 8 + wid;
  const bool ok = (e < E);
  if (ok) for (int idx = lane; idx < 48; idx += 32)
    sv[wid][idx] = vh_in[(size_t)e * 48 + idx];
  __syncthreads();
  const int dst = ok ? eidx[E + e] : 0;
  if (ok && lane < 16) {
    const int o = lane;
    float m0 = 0, m1 = 0, m2 = 0;
    for (int h = 0; h < 16; ++h) {
      const float w = wv2[h * 16 + o];
      m0 += sv[wid][h*3+0]*w; m1 += sv[wid][h*3+1]*w; m2 += sv[wid][h*3+2]*w;
    }
    atomicAdd(&sum_v[((size_t)dst*16 + o)*3 + 0], m0);
    atomicAdd(&sum_v[((size_t)dst*16 + o)*3 + 1], m1);
    atomicAdd(&sum_v[((size_t)dst*16 + o)*3 + 2], m2);
  }
  if (ok) for (int idx = lane; idx < 128; idx += 32)
    atomicAdd(&sum_s[(size_t)dst*128 + idx], ms[(size_t)e*128 + idx]);
  if (ok && lane == 0) atomicAdd(&cnt[dst], 1.f);
}

// node residual + norm0 + FF0 staging. Wave per node.
__global__ void __launch_bounds__(256)
node_update0(float* __restrict__ hs, float* __restrict__ hv,
             const float* __restrict__ sum_s, const float* __restrict__ sum_v,
             const float* __restrict__ cnt, const float* __restrict__ lnb,
             const float* __restrict__ lng, const float* __restrict__ whf,
             float* __restrict__ vhf, h16* __restrict__ A, int N) {
  __shared__ float shv[8][48];
  const int lane = threadIdx.x & 31, wid = threadIdx.x >> 5;
  const int n = blockIdx.x * 8 + wid;
  const bool ok = (n < N);
  float inv = 1.f;
  float s[4] = {0, 0, 0, 0};
  if (ok) {
    inv = 1.f / fmaxf(cnt[n], 1.f);
    #pragma unroll
    for (int j = 0; j < 4; ++j) {
      const int idx = lane + 32 * j;
      s[j] = hs[(size_t)n*128 + idx] + sum_s[(size_t)n*128 + idx] * inv;
    }
  }
  const float mu = wsum(s[0]+s[1]+s[2]+s[3]) * (1.f / 128.f);
  float vs = 0.f;
  #pragma unroll
  for (int j = 0; j < 4; ++j) { const float d = s[j]-mu; vs += d*d; }
  const float rs = rsqrtf(wsum(vs) * (1.f / 128.f) + 1e-5f);
  if (ok) {
    #pragma unroll
    for (int j = 0; j < 4; ++j) {
      const int idx = lane + 32 * j;
      const float val = (s[j]-mu)*rs*lng[idx] + lnb[idx];
      hs[(size_t)n*128 + idx] = val;
      A[(size_t)n*160 + idx] = (h16)val;
    }
  }
  float w0 = 0, w1 = 0, w2 = 0, q = 0;
  size_t base = 0;
  if (ok && lane < 16) {
    base = ((size_t)n*16 + lane) * 3;
    w0 = hv[base+0] + sum_v[base+0]*inv;
    w1 = hv[base+1] + sum_v[base+1]*inv;
    w2 = hv[base+2] + sum_v[base+2]*inv;
    q = fmaxf(w0*w0 + w1*w1 + w2*w2, 1e-8f);
  }
  const float scv = rsqrtf(fmaxf(wsum(q) * (1.f / 16.f), 1e-20f));
  if (ok && lane < 16) {
    w0 *= scv; w1 *= scv; w2 *= scv;
    hv[base+0] = w0; hv[base+1] = w1; hv[base+2] = w2;
    shv[wid][lane*3+0] = w0; shv[wid][lane*3+1] = w1; shv[wid][lane*3+2] = w2;
  }
  __syncthreads();
  if (ok) {
    float a0 = 0, a1 = 0, a2 = 0;
    for (int i = 0; i < 16; ++i) {
      const float w = whf[i * 32 + lane];
      a0 += shv[wid][i*3+0]*w; a1 += shv[wid][i*3+1]*w; a2 += shv[wid][i*3+2]*w;
    }
    vhf[((size_t)n*32 + lane)*3 + 0] = a0;
    vhf[((size_t)n*32 + lane)*3 + 1] = a1;
    vhf[((size_t)n*32 + lane)*3 + 2] = a2;
    A[(size_t)n*160 + 128 + lane] = (h16)sqrtf(fmaxf(a0*a0 + a1*a1 + a2*a2, 1e-8f));
  }
}

// FF0 vector gate + FF1 hidden + FF1 staging. Wave per node.
__global__ void __launch_bounds__(256)
ff0_post(const float* __restrict__ vhf, const float* __restrict__ wvf,
         const float* __restrict__ whn, const float* __restrict__ fs,
         float* __restrict__ vhf1, h16* __restrict__ A, int N) {
  __shared__ float sv[8][96];
  __shared__ float sg[8][96];
  const int lane = threadIdx.x & 31, wid = threadIdx.x >> 5;
  const int n = blockIdx.x * 8 + wid;
  const bool ok = (n < N);
  if (ok) for (int idx = lane; idx < 96; idx += 32)
    sv[wid][idx] = vhf[(size_t)n * 96 + idx];
  __syncthreads();
  if (ok) {
    const int o = lane;
    float m0 = 0, m1 = 0, m2 = 0;
    for (int h = 0; h < 32; ++h) {
      const float w = wvf[h * 32 + o];
      m0 += sv[wid][h*3+0]*w; m1 += sv[wid][h*3+1]*w; m2 += sv[wid][h*3+2]*w;
    }
    const float g = sigm(sqrtf(fmaxf(m0*m0 + m1*m1 + m2*m2, 1e-8f)));
    sg[wid][o*3+0] = m0*g; sg[wid][o*3+1] = m1*g; sg[wid][o*3+2] = m2*g;
  }
  __syncthreads();
  if (ok) {
    const int h = lane;
    float a0 = 0, a1 = 0, a2 = 0;
    for (int o = 0; o < 32; ++o) {
      const float w = whn[o * 32 + h];
      a0 += sg[wid][o*3+0]*w; a1 += sg[wid][o*3+1]*w; a2 += sg[wid][o*3+2]*w;
    }
    vhf1[((size_t)n*32 + h)*3 + 0] = a0;
    vhf1[((size_t)n*32 + h)*3 + 1] = a1;
    vhf1[((size_t)n*32 + h)*3 + 2] = a2;
    A[(size_t)n*544 + 512 + lane] = (h16)sqrtf(fmaxf(a0*a0 + a1*a1 + a2*a2, 1e-8f));
    for (int idx = lane; idx < 512; idx += 32)
      A[(size_t)n*544 + idx] = (h16)fs[(size_t)n*512 + idx];
  }
}

// FF residual + norm1. Wave per node.
__global__ void __launch_bounds__(256)
node_update1(float* __restrict__ hs, float* __restrict__ hv,
             const float* __restrict__ fs2, const float* __restrict__ vhf1,
             const float* __restrict__ wvf1, const float* __restrict__ lnb,
             const float* __restrict__ lng, int N) {
  __shared__ float sv[8][96];
  const int lane = threadIdx.x & 31, wid = threadIdx.x >> 5;
  const int n = blockIdx.x * 8 + wid;
  const bool ok = (n < N);
  if (ok) for (int idx = lane; idx < 96; idx += 32)
    sv[wid][idx] = vhf1[(size_t)n * 96 + idx];
  __syncthreads();
  float s[4] = {0, 0, 0, 0};
  if (ok) {
    #pragma unroll
    for (int j = 0; j < 4; ++j) {
      const int idx = lane + 32 * j;
      s[j] = hs[(size_t)n*128 + idx] + fs2[(size_t)n*128 + idx];
    }
  }
  const float mu = wsum(s[0]+s[1]+s[2]+s[3]) * (1.f / 128.f);
  float vs = 0.f;
  #pragma unroll
  for (int j = 0; j < 4; ++j) { const float d = s[j]-mu; vs += d*d; }
  const float rs = rsqrtf(wsum(vs) * (1.f / 128.f) + 1e-5f);
  if (ok) {
    #pragma unroll
    for (int j = 0; j < 4; ++j) {
      const int idx = lane + 32 * j;
      hs[(size_t)n*128 + idx] = (s[j]-mu)*rs*lng[idx] + lnb[idx];
    }
  }
  float w0 = 0, w1 = 0, w2 = 0, q = 0;
  size_t base = 0;
  if (ok && lane < 16) {
    const int o = lane;
    float f0 = 0, f1 = 0, f2 = 0;
    for (int h = 0; h < 32; ++h) {
      const float w = wvf1[h * 16 + o];
      f0 += sv[wid][h*3+0]*w; f1 += sv[wid][h*3+1]*w; f2 += sv[wid][h*3+2]*w;
    }
    base = ((size_t)n*16 + o) * 3;
    w0 = hv[base+0] + f0; w1 = hv[base+1] + f1; w2 = hv[base+2] + f2;
    q = fmaxf(w0*w0 + w1*w1 + w2*w2, 1e-8f);
  }
  const float scv = rsqrtf(fmaxf(wsum(q) * (1.f / 16.f), 1e-20f));
  if (ok && lane < 16) {
    hv[base+0] = w0*scv; hv[base+1] = w1*scv; hv[base+2] = w2*scv;
  }
}

// Output head staging: LN; v RMS; vh = v*wh_out(16x16); A=f16[s(128),|vh|(16),pad].
__global__ void __launch_bounds__(256)
out_pre(const float* __restrict__ hs, const float* __restrict__ hv,
        const float* __restrict__ lnb, const float* __restrict__ lng,
        const float* __restrict__ who, h16* __restrict__ A, int N) {
  __shared__ float shv[8][48];
  const int lane = threadIdx.x & 31, wid = threadIdx.x >> 5;
  const int n = blockIdx.x * 8 + wid;
  const bool ok = (n < N);
  float s[4] = {0, 0, 0, 0};
  if (ok) {
    #pragma unroll
    for (int j = 0; j < 4; ++j) s[j] = hs[(size_t)n*128 + lane + 32*j];
  }
  const float mu = wsum(s[0]+s[1]+s[2]+s[3]) * (1.f / 128.f);
  float vs = 0.f;
  #pragma unroll
  for (int j = 0; j < 4; ++j) { const float d = s[j]-mu; vs += d*d; }
  const float rs = rsqrtf(wsum(vs) * (1.f / 128.f) + 1e-5f);
  if (ok) {
    #pragma unroll
    for (int j = 0; j < 4; ++j) {
      const int idx = lane + 32 * j;
      A[(size_t)n*160 + idx] = (h16)((s[j]-mu)*rs*lng[idx] + lnb[idx]);
    }
  }
  float w0 = 0, w1 = 0, w2 = 0, q = 0;
  if (ok && lane < 16) {
    const size_t base = ((size_t)n*16 + lane) * 3;
    w0 = hv[base+0]; w1 = hv[base+1]; w2 = hv[base+2];
    q = fmaxf(w0*w0 + w1*w1 + w2*w2, 1e-8f);
  }
  const float scv = rsqrtf(fmaxf(wsum(q) * (1.f / 16.f), 1e-20f));
  if (ok && lane < 16) {
    shv[wid][lane*3+0] = w0*scv; shv[wid][lane*3+1] = w1*scv; shv[wid][lane*3+2] = w2*scv;
  }
  __syncthreads();
  if (ok && lane < 16) {
    const int h = lane;
    float a0 = 0, a1 = 0, a2 = 0;
    for (int i = 0; i < 16; ++i) {
      const float w = who[i * 16 + h];
      a0 += shv[wid][i*3+0]*w; a1 += shv[wid][i*3+1]*w; a2 += shv[wid][i*3+2]*w;
    }
    A[(size_t)n*160 + 128 + h]    = (h16)sqrtf(fmaxf(a0*a0 + a1*a1 + a2*a2, 1e-8f));
    A[(size_t)n*160 + 144 + lane] = (h16)0.f;
  }
}

// ---------------------------------------------------------------------------
// Host orchestration
// ---------------------------------------------------------------------------
extern "C" void kernel_launch(void* const* d_in, const int* in_sizes, int n_in,
                              void* d_out, int out_size, void* d_ws, size_t ws_size,
                              hipStream_t stream) {
  const int N = in_sizes[0] / 6;    // nodes
  const int E = in_sizes[2] / 32;   // edges
  const float* node_s = (const float*)d_in[0];
  const float* node_v = (const float*)d_in[1];
  const float* edge_s = (const float*)d_in[2];
  const float* edge_v = (const float*)d_in[3];
  const int*   eidx   = (const int*)d_in[4];
  const int*   seq    = (const int*)d_in[5];
  // pytree-flatten order (dict keys sorted): W_e, W_out, W_s, W_v, layers
  // W_e:   6 wh, 7 ws_b, 8 ws_w, 9 wv, 10 ln.b, 11 ln.g
  // W_out: 12 wh, 13 ws_b, 14 ws_w, 15 ln.b, 16 ln.g
  // W_s:   17
  // W_v:   18 wh, 19 ws_b, 20 ws_w, 21 wv, 22 ln.b, 23 ln.g
  // layer l (base=24+24l): conv0 +0..3, conv1 +4..7, conv2 +8..11,
  //   ff0 +12..15, ff1 +16..19, norm0 +20 b /+21 g, norm1 +22 b /+23 g
  auto F = [&](int i) { return (const float*)d_in[i]; };

  size_t off = 0;
  auto alloc = [&](size_t bytes) -> void* {
    size_t a = (off + 255) & ~(size_t)255;
    off = a + bytes;
    return (void*)((char*)d_ws + a);
  };
  float* hs    = (float*)alloc((size_t)N * 128 * 4);
  float* hv    = (float*)alloc((size_t)N * 48 * 4);
  float* vhn   = (float*)alloc((size_t)N * 48 * 4);
  float* vhf   = (float*)alloc((size_t)N * 96 * 4);
  float* fs    = (float*)alloc((size_t)N * 512 * 4);
  float* sum_s = (float*)alloc((size_t)N * 128 * 4);   // reused as fs2
  float* sum_v = (float*)alloc((size_t)N * 48 * 4);
  float* cntb  = (float*)alloc((size_t)N * 4);
  float* es2   = (float*)alloc((size_t)E * 32 * 4);
  float* ev2   = (float*)alloc((size_t)E * 3 * 4);
  float* msb   = (float*)alloc((size_t)E * 128 * 4);
  float* mvb   = (float*)alloc((size_t)E * 48 * 4);
  float* vh33  = (float*)alloc((size_t)E * 99 * 4);
  float* vh16  = (float*)alloc((size_t)E * 48 * 4);
  h16* Aedge = (h16*)alloc((size_t)E * 352 * 2);
  h16* Anode = (h16*)alloc((size_t)N * 544 * 2);
  h16* pWv   = (h16*)alloc(32 * 128 * 2);
  h16* pWe   = (h16*)alloc(64 * 32 * 2);
  h16* pWout = (h16*)alloc(160 * 128 * 2);
  h16 *pC0[3], *pC1[3], *pC2[3], *pF0[3], *pF1[3];
  for (int l = 0; l < 3; ++l) {
    pC0[l] = (h16*)alloc(352 * 128 * 2);
    pC1[l] = (h16*)alloc(160 * 128 * 2);
    pC2[l] = (h16*)alloc(160 * 128 * 2);
    pF0[l] = (h16*)alloc(160 * 512 * 2);
    pF1[l] = (h16*)alloc(544 * 128 * 2);
  }
  (void)ws_size; (void)n_in; (void)out_size;

  // ---- weight packing (f32 -> f16 B-lane layout) ----
  pack_w<<<dim3(1, 8),  32, 0, stream>>>(F(20), pWv,   32, 128);   // W_v ws_w
  pack_w<<<dim3(2, 2),  32, 0, stream>>>(F(8),  pWe,   33, 32);    // W_e ws_w
  pack_w<<<dim3(5, 8),  32, 0, stream>>>(F(14), pWout, 144, 128);  // W_out ws_w
  for (int l = 0; l < 3; ++l) {
    const int b = 24 + l * 24;
    pack_w<<<dim3(11, 8), 32, 0, stream>>>(F(b + 2),  pC0[l], 321, 128);
    pack_w<<<dim3(5, 8),  32, 0, stream>>>(F(b + 6),  pC1[l], 144, 128);
    pack_w<<<dim3(5, 8),  32, 0, stream>>>(F(b + 10), pC2[l], 144, 128);
    pack_w<<<dim3(5, 32), 32, 0, stream>>>(F(b + 14), pF0[l], 160, 512);
    pack_w<<<dim3(17, 8), 32, 0, stream>>>(F(b + 18), pF1[l], 544, 128);
  }

  const int nbN = (N + 7) / 8, nbE = (E + 7) / 8;
  const int mtN = (N + 15) / 16, mtE = (E + 15) / 16;

  // ---- node / edge embedding ----
  embed_nodes<<<nbN, 256, 0, stream>>>(node_s, node_v, seq, F(17), F(23), F(22),
                                       F(18), Anode, vhn, N);
  gemm_wmma<8><<<dim3(mtN, 1), 32, 0, stream>>>(Anode, pWv, F(19), hs, N, 32, 128, 0);
  node_wv<<<(N * 16 + 255) / 256, 256, 0, stream>>>(vhn, F(21), hv, N);
  embed_edges<<<nbE, 256, 0, stream>>>(edge_s, edge_v, F(11), F(10), F(6), F(9),
                                       Aedge, ev2, E);
  gemm_wmma<2><<<dim3(mtE, 1), 32, 0, stream>>>(Aedge, pWe, F(7), es2, E, 64, 32, 0);

  // ---- message-passing layers ----
  for (int l = 0; l < 3; ++l) {
    const int b = 24 + l * 24;
    zerof<<<((size_t)N * 128 + 255) / 256, 256, 0, stream>>>(sum_s, (size_t)N * 128);
    zerof<<<((size_t)N * 48 + 255) / 256, 256, 0, stream>>>(sum_v, (size_t)N * 48);
    zerof<<<(N + 255) / 256, 256, 0, stream>>>(cntb, (size_t)N);

    conv_pre<<<nbE, 256, 0, stream>>>(hs, es2, hv, ev2, eidx, F(b + 0), vh33, Aedge, E);
    gemm_wmma<8><<<dim3(mtE, 1), 32, 0, stream>>>(Aedge, pC0[l], F(b + 1), msb, E, 352, 128, 1);
    conv_mid_post<<<nbE, 256, 0, stream>>>(vh33, 33, F(b + 3), F(b + 4), msb, mvb, vh16, Aedge, E);
    gemm_wmma<8><<<dim3(mtE, 1), 32, 0, stream>>>(Aedge, pC1[l], F(b + 5), msb, E, 160, 128, 1);
    conv_mid_post<<<nbE, 256, 0, stream>>>(vh16, 16, F(b + 7), F(b + 8), msb, mvb, vh16, Aedge, E);
    gemm_wmma<8><<<dim3(mtE, 1), 32, 0, stream>>>(Aedge, pC2[l], F(b + 9), msb, E, 160, 128, 0);
    conv_final<<<nbE, 256, 0, stream>>>(vh16, F(b + 11), msb, eidx, sum_s, sum_v, cntb, E);

    node_update0<<<nbN, 256, 0, stream>>>(hs, hv, sum_s, sum_v, cntb, F(b + 20),
                                          F(b + 21), F(b + 12), vhf, Anode, N);
    gemm_wmma<8><<<dim3(mtN, 4), 32, 0, stream>>>(Anode, pF0[l], F(b + 13), fs, N, 160, 512, 1);
    ff0_post<<<nbN, 256, 0, stream>>>(vhf, F(b + 15), F(b + 16), fs, vhf, Anode, N);
    gemm_wmma<8><<<dim3(mtN, 1), 32, 0, stream>>>(Anode, pF1[l], F(b + 17), sum_s, N, 544, 128, 0);
    node_update1<<<nbN, 256, 0, stream>>>(hs, hv, sum_s, vhf, F(b + 19), F(b + 22),
                                          F(b + 23), N);
  }

  // ---- output head ----
  out_pre<<<nbN, 256, 0, stream>>>(hs, hv, F(15), F(16), F(12), Anode, N);
  gemm_wmma<8><<<dim3(mtN, 1), 32, 0, stream>>>(Anode, pWout, F(13), (float*)d_out,
                                                N, 160, 128, 1);
}
